// LangevinOscillator_26603027431567
// MI455X (gfx1250) — matrix operand order repaired and checked by reading, not compiled, not run on hardware
//
#include <hip/hip_runtime.h>

typedef __attribute__((ext_vector_type(16))) _Float16 v16h;
typedef __attribute__((ext_vector_type(8)))  float    v8f;

namespace {
constexpr int   BDIM = 65536;
constexpr int   NDIM = 1024;
constexpr float DT_C = 0.01f;
constexpr float NSC  = 0.01f;   // NOISE_STD * sqrt(DT) = 0.1 * 0.1

// fragment <-> raw-bits view for WMMA operands
union Frag {
    v16h          v;
    _Float16      h[16];
    unsigned int  w[8];
};

// CDNA5 async copy: 16 bytes global -> LDS per lane, tracked by ASYNCcnt.
__device__ __forceinline__ void async_cp16(const _Float16* lds_dst, const _Float16* gsrc) {
    // Low 32 bits of the generic shared-aperture address are the LDS byte offset.
    unsigned           l = (unsigned)(unsigned long long)(uintptr_t)lds_dst;
    unsigned long long g = (unsigned long long)(uintptr_t)gsrc;
    asm volatile("global_load_async_to_lds_b128 %0, %1, off"
                 :: "v"(l), "v"(g) : "memory");
}
__device__ __forceinline__ void wait_async0() {
    asm volatile("s_wait_asynccnt 0" ::: "memory");
}
} // namespace

// ---------------- precompute kernels (run once per launch) ----------------

__global__ __launch_bounds__(256)
void precompute_sincos(const float* __restrict__ theta,
                       _Float16* __restrict__ sArr,
                       _Float16* __restrict__ cArr) {
    const size_t i4 = ((size_t)blockIdx.x * 256 + threadIdx.x) * 4;
    const float4 t = *(const float4*)(theta + i4);
    float s0, c0, s1, c1, s2, c2, s3, c3;
    __sincosf(t.x, &s0, &c0);
    __sincosf(t.y, &s1, &c1);
    __sincosf(t.z, &s2, &c2);
    __sincosf(t.w, &s3, &c3);
    union { _Float16 h[4]; uint2 u; } S, C;
    S.h[0] = (_Float16)s0; S.h[1] = (_Float16)s1; S.h[2] = (_Float16)s2; S.h[3] = (_Float16)s3;
    C.h[0] = (_Float16)c0; C.h[1] = (_Float16)c1; C.h[2] = (_Float16)c2; C.h[3] = (_Float16)c3;
    *(uint2*)(sArr + i4) = S.u;
    *(uint2*)(cArr + i4) = C.u;
}

__global__ __launch_bounds__(256)
void precompute_kf16(const float* __restrict__ Kmat, _Float16* __restrict__ Kh) {
    const size_t i4 = ((size_t)blockIdx.x * 256 + threadIdx.x) * 4;
    const float4 k = *(const float4*)(Kmat + i4);
    union { _Float16 h[4]; uint2 u; } H;
    H.h[0] = (_Float16)k.x; H.h[1] = (_Float16)k.y;
    H.h[2] = (_Float16)k.z; H.h[3] = (_Float16)k.w;
    *(uint2*)(Kh + i4) = H.u;
}

// ---------------- fused WMMA GEMM + Langevin update ----------------

template <bool PRE>
__global__ __launch_bounds__(256)
void kuramoto_gemm(const float* __restrict__ theta,
                   const float* __restrict__ noise,
                   const float* __restrict__ omega,
                   const float* __restrict__ Kmat,
                   const _Float16* __restrict__ sArr,
                   const _Float16* __restrict__ cArr,
                   const _Float16* __restrict__ Khalf,
                   float* __restrict__ out) {
    constexpr int NB = PRE ? 2 : 1;            // double-buffer only on async path
    __shared__ _Float16 sAs[NB][128 * 32];
    __shared__ _Float16 sAc[NB][128 * 32];
    __shared__ _Float16 sBk[NB][64 * 32];

    const int tid  = threadIdx.x;
    const int wave = tid >> 5;                 // 16-row strip of the 128-row tile
    const int lane = tid & 31;
    const int half = lane >> 4;
    const int col  = lane & 15;
    const int b0   = blockIdx.y * 128;
    const int i0   = blockIdx.x * 64;

    v8f accS[4], accC[4];
    const v8f vzero = {0.f, 0.f, 0.f, 0.f, 0.f, 0.f, 0.f, 0.f};
#pragma unroll
    for (int nt = 0; nt < 4; ++nt) { accS[nt] = vzero; accC[nt] = vzero; }

    // async staging: 5 x 16B per thread per k-chunk (2 sin + 2 cos + 1 K)
    auto stage_async = [&](int buf, int k0) {
#pragma unroll
        for (int q = 0; q < 2; ++q) {
            const int id  = q * 256 + tid;     // 0..511 16B chunks (128 rows x 4 segs)
            const int r   = id >> 2;
            const int seg = id & 3;
            async_cp16(&sAs[buf][r * 32 + seg * 8],
                       sArr + (size_t)(b0 + r) * NDIM + k0 + seg * 8);
            async_cp16(&sAc[buf][r * 32 + seg * 8],
                       cArr + (size_t)(b0 + r) * NDIM + k0 + seg * 8);
        }
        {
            const int r   = tid >> 2;          // 0..63
            const int seg = tid & 3;
            async_cp16(&sBk[buf][r * 32 + seg * 8],
                       Khalf + (size_t)(i0 + r) * NDIM + k0 + seg * 8);
        }
    };

    // fallback staging: fp32 loads + sincos + f16 convert in-kernel
    auto stage_sincos = [&](int k0) {
#pragma unroll
        for (int q = 0; q < 4; ++q) {
            const int idx = q * 256 + tid;
            const int r   = idx >> 3;
            const int cv  = idx & 7;
            const float4 tv = *(const float4*)(theta + (size_t)(b0 + r) * NDIM + k0 + cv * 4);
            float s0, c0, s1, c1, s2, c2, s3, c3;
            __sincosf(tv.x, &s0, &c0);
            __sincosf(tv.y, &s1, &c1);
            __sincosf(tv.z, &s2, &c2);
            __sincosf(tv.w, &s3, &c3);
            _Float16* ps = &sAs[0][r * 32 + cv * 4];
            _Float16* pc = &sAc[0][r * 32 + cv * 4];
            ps[0] = (_Float16)s0; ps[1] = (_Float16)s1; ps[2] = (_Float16)s2; ps[3] = (_Float16)s3;
            pc[0] = (_Float16)c0; pc[1] = (_Float16)c1; pc[2] = (_Float16)c2; pc[3] = (_Float16)c3;
        }
#pragma unroll
        for (int q = 0; q < 2; ++q) {
            const int idx = q * 256 + tid;
            const int r   = idx >> 3;
            const int cv  = idx & 7;
            const float4 kv = *(const float4*)(Kmat + (size_t)(i0 + r) * NDIM + k0 + cv * 4);
            _Float16* pb = &sBk[0][r * 32 + cv * 4];
            pb[0] = (_Float16)kv.x; pb[1] = (_Float16)kv.y;
            pb[2] = (_Float16)kv.z; pb[3] = (_Float16)kv.w;
        }
    };

    if (PRE) {
        stage_async(0, 0);
        wait_async0();
        __syncthreads();
    }

    for (int k0 = 0; k0 < NDIM; k0 += 32) {
        int cur = 0;
        if (PRE) {
            cur = (k0 >> 5) & 1;
            if (k0 + 32 < NDIM) stage_async(cur ^ 1, k0 + 32);   // prefetch next chunk
        } else {
            __syncthreads();
            stage_sincos(k0);
            __syncthreads();
        }

        // A fragments (16x32 f16: M = lane%16; K = half*8+0..7, 16+half*8+0..7)
        Frag aS, aC;
        {
            const _Float16* p = &sAs[cur][(wave * 16 + col) * 32 + half * 8];
            *(uint4*)&aS.w[0] = *(const uint4*)(const void*)p;
            *(uint4*)&aS.w[4] = *(const uint4*)(const void*)(p + 16);
            const _Float16* q = &sAc[cur][(wave * 16 + col) * 32 + half * 8];
            *(uint4*)&aC.w[0] = *(const uint4*)(const void*)q;
            *(uint4*)&aC.w[4] = *(const uint4*)(const void*)(q + 16);
        }

        // B fragments (32x16 f16: N = lane%16; K = half*8+0..7, 16+half*8+0..7) + 8 WMMAs
#pragma unroll
        for (int nt = 0; nt < 4; ++nt) {
            Frag bF;
            const _Float16* pb = &sBk[cur][(nt * 16 + col) * 32 + half * 8];
            *(uint4*)&bF.w[0] = *(const uint4*)(const void*)pb;
            *(uint4*)&bF.w[4] = *(const uint4*)(const void*)(pb + 16);
            accS[nt] = __builtin_amdgcn_wmma_f32_16x16x32_f16(
                false, aS.v, false, bF.v, (short)0, accS[nt], false, false);
            accC[nt] = __builtin_amdgcn_wmma_f32_16x16x32_f16(
                false, aC.v, false, bF.v, (short)0, accC[nt], false, false);
        }

        if (PRE) {
            wait_async0();      // prefetch landed in LDS
            __syncthreads();    // publish to all waves; prior buffer reads done
        }
    }

    // epilogue: C/D layout (VGPR m, half -> row m+8*half; lane%16 -> col)
#pragma unroll
    for (int nt = 0; nt < 4; ++nt) {
        const int   i  = i0 + nt * 16 + col;
        const float om = omega[i];
#pragma unroll
        for (int m = 0; m < 8; ++m) {
            const int    b   = b0 + wave * 16 + half * 8 + m;
            const size_t idx = (size_t)b * NDIM + i;
            const float  th  = theta[idx];
            float s, c;
            __sincosf(th, &s, &c);
            const float coupling = c * accS[nt][m] - s * accC[nt][m];
            out[idx] = th + (om + coupling) * DT_C + noise[idx] * NSC;
        }
    }
}

extern "C" void kernel_launch(void* const* d_in, const int* in_sizes, int n_in,
                              void* d_out, int out_size, void* d_ws, size_t ws_size,
                              hipStream_t stream) {
    (void)in_sizes; (void)n_in; (void)out_size;
    const float* theta = (const float*)d_in[0];
    const float* noise = (const float*)d_in[1];
    const float* omega = (const float*)d_in[2];
    const float* Kmat  = (const float*)d_in[3];
    float* out = (float*)d_out;

    const size_t SC_BYTES = (size_t)BDIM * NDIM * sizeof(_Float16);   // 128 MB each
    const size_t K_BYTES  = (size_t)NDIM * NDIM * sizeof(_Float16);   // 2 MB
    const dim3 grid(NDIM / 64, BDIM / 128);    // 16 x 512
    const dim3 block(256);                     // 8 wave32s

    if (ws_size >= 2 * SC_BYTES + K_BYTES) {
        _Float16* sArr = (_Float16*)d_ws;
        _Float16* cArr = (_Float16*)((char*)d_ws + SC_BYTES);
        _Float16* Kh   = (_Float16*)((char*)d_ws + 2 * SC_BYTES);
        hipLaunchKernelGGL(precompute_sincos, dim3((BDIM * (size_t)NDIM) / 1024), block, 0, stream,
                           theta, sArr, cArr);
        hipLaunchKernelGGL(precompute_kf16, dim3((NDIM * (size_t)NDIM) / 1024), block, 0, stream,
                           Kmat, Kh);
        hipLaunchKernelGGL((kuramoto_gemm<true>), grid, block, 0, stream,
                           theta, noise, omega, Kmat, sArr, cArr, Kh, out);
    } else {
        hipLaunchKernelGGL((kuramoto_gemm<false>), grid, block, 0, stream,
                           theta, noise, omega, Kmat,
                           (const _Float16*)nullptr, (const _Float16*)nullptr,
                           (const _Float16*)nullptr, out);
    }
}
